// STPatchMaskFormer_47974784696504
// MI455X (gfx1250) — compile-verified, matrix operand order checked by reading, not run on hardware
//
#include <hip/hip_runtime.h>
#include <hip/hip_bf16.h>

typedef __attribute__((ext_vector_type(16))) __bf16          v16bf;
typedef __attribute__((ext_vector_type(8)))  float           v8f;
typedef __attribute__((ext_vector_type(8)))  unsigned short  u16x8;

union FragA { u16x8 h[2]; v16bf bf; };
union Acc8  { v8f v; float f[8]; };
union U8x8  { u16x8 v; unsigned short s[8]; };

__device__ __forceinline__ unsigned short f2bf(float f) {
  unsigned int u = __float_as_uint(f);
  u += 0x7fffu + ((u >> 16) & 1u);  // round-to-nearest-even
  return (unsigned short)(u >> 16);
}

__device__ __forceinline__ v8f wmma_bf16(v16bf a, v16bf b, v8f c) {
  return __builtin_amdgcn_wmma_f32_16x16x32_bf16(false, a, false, b, (short)0, c,
                                                 false, false);
}

// ------------------------------------------------------------- converters
__global__ void cvt_f32_bf16_kernel(const float* __restrict__ src,
                                    unsigned short* __restrict__ dst, int n) {
  for (int i = blockIdx.x * blockDim.x + threadIdx.x; i < n;
       i += gridDim.x * blockDim.x)
    dst[i] = f2bf(src[i]);
}

// zero-padded 2D convert: dst[dstH,dstW] <- src[srcH,srcW] (pad with 0)
__global__ void cvt_pad2d_kernel(const float* __restrict__ src,
                                 unsigned short* __restrict__ dst, int srcH,
                                 int srcW, int dstH, int dstW) {
  int total = dstH * dstW;
  for (int i = blockIdx.x * blockDim.x + threadIdx.x; i < total;
       i += gridDim.x * blockDim.x) {
    int r = i / dstW, c = i - r * dstW;
    dst[i] = (r < srcH && c < srcW) ? f2bf(src[(size_t)r * srcW + c])
                                    : (unsigned short)0;
  }
}

// ------------------------------------------------------------- WMMA GEMM
// C[M,N] = A[M,K](bf16,lda) x B[K,N](bf16,ldb) (+bias)(+resid)(+relu)
// Preconditions (guaranteed by launcher via padding): M % 256 == 0,
// K % 32 == 0, B buffer valid for all n in [n0, n0+64). Only the epilogue
// guards col < Nout. Block = 256 threads = 8 waves; tile 256(M) x 64(N);
// double-buffered LDS, 8 WMMA per wave per K-step.
__global__ __launch_bounds__(256)
void gemm_bf16_kernel(const unsigned short* __restrict__ A, int lda,
                      const unsigned short* __restrict__ B, int ldb,
                      const float* __restrict__ bias,
                      const float* __restrict__ resid,
                      float* __restrict__ outF,
                      unsigned short* __restrict__ outB,
                      int Nout, int K, int relu) {
  __shared__ __align__(16) unsigned short As[2][256 * 40];  // 32 cols + pad
  __shared__ __align__(16) unsigned short Bs[2][64 * 40];   // transposed + pad

  const int tid = threadIdx.x;
  const size_t m0 = (size_t)blockIdx.x * 256;
  const int n0 = blockIdx.y * 64;
  const int wv = tid >> 5, lane = tid & 31, lm = lane & 15, hi = lane >> 4;
  const int c0 = hi ? 8 : 0, c1 = c0 + 16;
  const int br = tid >> 3, bc = (tid & 7) * 8;

  Acc8 acc[8];
#pragma unroll
  for (int t = 0; t < 8; t++)
#pragma unroll
    for (int r = 0; r < 8; r++) acc[t].f[r] = 0.f;

  const unsigned short* aP = A + (m0 + tid) * (size_t)lda;  // this thread's A row

  u16x8 aReg[4];
  U8x8 bReg;
#pragma unroll
  for (int q = 0; q < 4; q++) aReg[q] = *(const u16x8*)(aP + q * 8);
  bReg.v = *(const u16x8*)(B + (size_t)br * ldb + n0 + bc);
#pragma unroll
  for (int q = 0; q < 4; q++) *(u16x8*)(&As[0][tid * 40 + q * 8]) = aReg[q];
#pragma unroll
  for (int j = 0; j < 8; j++) Bs[0][(bc + j) * 40 + br] = bReg.s[j];
  __syncthreads();

  const int nK = K >> 5;
  for (int ks = 0; ks < nK; ks++) {
    const int cur = ks & 1;
    const bool more = (ks + 1) < nK;
    if (more) {  // prefetch next tile global -> regs (hidden behind WMMAs)
      int k0 = (ks + 1) << 5;
#pragma unroll
      for (int q = 0; q < 4; q++)
        aReg[q] = *(const u16x8*)(aP + k0 + q * 8);
      bReg.v = *(const u16x8*)(B + (size_t)(k0 + br) * ldb + n0 + bc);
    }
    const unsigned short* as = &As[cur][0];
    const unsigned short* bs = &Bs[cur][0];
    FragA a0, a1;
    a0.h[0] = *(const u16x8*)(as + (wv * 32 + lm) * 40 + c0);
    a0.h[1] = *(const u16x8*)(as + (wv * 32 + lm) * 40 + c1);
    a1.h[0] = *(const u16x8*)(as + (wv * 32 + 16 + lm) * 40 + c0);
    a1.h[1] = *(const u16x8*)(as + (wv * 32 + 16 + lm) * 40 + c1);
    FragA bf[4];
#pragma unroll
    for (int nt = 0; nt < 4; nt++) {
      bf[nt].h[0] = *(const u16x8*)(bs + (nt * 16 + lm) * 40 + c0);
      bf[nt].h[1] = *(const u16x8*)(bs + (nt * 16 + lm) * 40 + c1);
    }
#pragma unroll
    for (int nt = 0; nt < 4; nt++) {
      acc[nt].v = wmma_bf16(a0.bf, bf[nt].bf, acc[nt].v);
      acc[4 + nt].v = wmma_bf16(a1.bf, bf[nt].bf, acc[4 + nt].v);
    }
    if (more) {
      const int nxt = cur ^ 1;
#pragma unroll
      for (int q = 0; q < 4; q++)
        *(u16x8*)(&As[nxt][tid * 40 + q * 8]) = aReg[q];
#pragma unroll
      for (int j = 0; j < 8; j++) Bs[nxt][(bc + j) * 40 + br] = bReg.s[j];
    }
    __syncthreads();
  }

#pragma unroll
  for (int half = 0; half < 2; half++) {
#pragma unroll
    for (int nt = 0; nt < 4; nt++) {
#pragma unroll
      for (int r = 0; r < 8; r++) {
        size_t row = m0 + wv * 32 + half * 16 + (hi ? r + 8 : r);
        int col = n0 + nt * 16 + lm;
        if (col < Nout) {
          float v = acc[half * 4 + nt].f[r];
          if (bias) v += bias[col];
          if (resid) v += resid[row * Nout + col];
          if (relu) v = fmaxf(v, 0.f);
          size_t off = row * Nout + col;
          if (outF) outF[off] = v;
          if (outB) outB[off] = f2bf(v);
        }
      }
    }
  }
}

// ------------------------------------------------------------- LayerNorm (D=256)
__global__ __launch_bounds__(256)
void ln_kernel(const float* __restrict__ in, const float* __restrict__ g,
               const float* __restrict__ b, float* __restrict__ outF,
               unsigned short* __restrict__ outB, int rows) {
  int wv = threadIdx.x >> 5, lane = threadIdx.x & 31;
  int row = blockIdx.x * 8 + wv;
  if (row >= rows) return;
  const float* p = in + (size_t)row * 256;
  float x[8];
  float s = 0.f, s2 = 0.f;
#pragma unroll
  for (int i = 0; i < 8; i++) {
    x[i] = p[lane + i * 32];
    s += x[i];
    s2 += x[i] * x[i];
  }
#pragma unroll
  for (int off = 16; off > 0; off >>= 1) {
    s += __shfl_xor(s, off, 32);
    s2 += __shfl_xor(s2, off, 32);
  }
  float mu = s * (1.f / 256.f);
  float var = s2 * (1.f / 256.f) - mu * mu;  // jnp.var, ddof=0
  float inv = rsqrtf(var + 1e-5f);
#pragma unroll
  for (int i = 0; i < 8; i++) {
    int c = lane + i * 32;
    float y = (x[i] - mu) * inv * g[c] + b[c];
    outF[(size_t)row * 256 + c] = y;
    outB[(size_t)row * 256 + c] = f2bf(y);
  }
}

// ------------------------------------------------------------- CV + top-k mask
// one block per (b,v); thread p handles patch p. mask=1.0 -> keep (NOT in top-k)
__global__ __launch_bounds__(256)
void cv_mask_kernel(const float* __restrict__ x, float* __restrict__ maskOut,
                    int kSel) {
  __shared__ float cvs[256];
  int b = blockIdx.x >> 5, v = blockIdx.x & 31, p = threadIdx.x;
  const float* px = x + (((size_t)(b * 256 + p)) * 32 + v) * 16;
  float s = 0.f;
#pragma unroll
  for (int j = 0; j < 16; j++) s += px[j];
  float mu = s * (1.f / 16.f);
  float ss = 0.f;
#pragma unroll
  for (int j = 0; j < 16; j++) {
    float d = px[j] - mu;
    ss += d * d;
  }
  float cv = sqrtf(ss * (1.f / 15.f)) / (mu + 1e-6f);  // torch.std: ddof=1
  cvs[p] = cv;
  __syncthreads();
  int cnt = 0;
  for (int j = 0; j < 256; j++) {
    float o = cvs[j];
    cnt += (o > cv) || ((o == cv) && (j < p));  // stable top-k rank
  }
  maskOut[((size_t)(b * 256 + p)) * 32 + v] = (cnt < kSel) ? 0.f : 1.f;
}

// ------------------------------------------- mask-token substitute + permute
// (b,p,v) token order -> (b,v,p) decoder order
__global__ __launch_bounds__(256)
void mask_apply_kernel(const float* __restrict__ xf, const float* __restrict__ mask,
                       const float* __restrict__ mtok, float* __restrict__ tf,
                       unsigned short* __restrict__ tb) {
  int idx = blockIdx.x * 256 + threadIdx.x;  // 65536 blocks x 256
  int t = idx >> 8, c = idx & 255;
  int b = t >> 13, rem = t & 8191, p = rem >> 5, v = rem & 31;
  float val = (mask[t] != 0.f) ? xf[(size_t)t * 256 + c] : mtok[c];
  size_t dst = ((size_t)((b * 32 + v) * 256 + p)) * 256 + c;
  tf[dst] = val;
  tb[dst] = f2bf(val);
}

// ------------------------------------------------------------- fused attention
// one block per (seq, head); L=256 keys, dh=32 (one bf16 WMMA K-step for QK^T).
// Q/K fragments straight from global (L2-resident), V staged transposed in LDS.
__global__ __launch_bounds__(128)
void attn_kernel(const unsigned short* __restrict__ qkv,
                 unsigned short* __restrict__ o) {
  int seq = blockIdx.x >> 3;
  int h = blockIdx.x & 7;
  __shared__ __align__(16) unsigned short Vt[32 * 264];   // V^T [dh][key], padded
  __shared__ __align__(16) unsigned short Ps[4][16 * 264];  // probs, padded rows
  int tid = threadIdx.x;
#pragma unroll
  for (int rr = 0; rr < 2; rr++) {  // stage V transposed
    int row = tid * 2 + rr;  // key index
    const unsigned short* src =
        qkv + (size_t)(seq * 256 + row) * 768 + 512 + h * 32;
    U8x8 vv[4];
#pragma unroll
    for (int q = 0; q < 4; q++) vv[q].v = ((const u16x8*)src)[q];
#pragma unroll
    for (int c = 0; c < 32; c++) Vt[c * 264 + row] = vv[c >> 3].s[c & 7];
  }
  __syncthreads();

  int wv = tid >> 5, lane = tid & 31, lm = lane & 15, hi = lane >> 4;
  int c0 = hi ? 8 : 0, c1 = c0 + 16;
  const float scale = 0.17677669529663687f;  // 1/sqrt(32)
  const unsigned short* kbase = qkv + (size_t)(seq * 256) * 768 + 256 + h * 32;

  for (int rt = 0; rt < 4; rt++) {  // each wave: 4 row-tiles of 16 queries
    int tile = wv * 4 + rt;
    int qrow = tile * 16 + lm;
    const unsigned short* qsrc =
        qkv + (size_t)(seq * 256 + qrow) * 768 + h * 32;
    FragA qa;
    qa.h[0] = *(const u16x8*)(qsrc + c0);
    qa.h[1] = *(const u16x8*)(qsrc + c1);

    Acc8 sacc[16];
#pragma unroll
    for (int nt = 0; nt < 16; nt++)
#pragma unroll
      for (int r = 0; r < 8; r++) sacc[nt].f[r] = 0.f;

#pragma unroll
    for (int nt = 0; nt < 16; nt++) {  // S = Q K^T
      int n = nt * 16 + lm;
      FragA kb;
      kb.h[0] = *(const u16x8*)(kbase + (size_t)n * 768 + c0);
      kb.h[1] = *(const u16x8*)(kbase + (size_t)n * 768 + c1);
      sacc[nt].v = wmma_bf16(qa.bf, kb.bf, sacc[nt].v);
    }

    unsigned short* pp = &Ps[wv][0];
#pragma unroll
    for (int r = 0; r < 8; r++) {  // softmax over 256 keys
      float mv = -1e30f;
#pragma unroll
      for (int nt = 0; nt < 16; nt++) mv = fmaxf(mv, sacc[nt].f[r]);
#pragma unroll
      for (int off = 8; off > 0; off >>= 1)
        mv = fmaxf(mv, __shfl_xor(mv, off, 16));
      float sum = 0.f;
#pragma unroll
      for (int nt = 0; nt < 16; nt++) {
        float pv = __expf((sacc[nt].f[r] - mv) * scale);
        sacc[nt].f[r] = pv;
        sum += pv;
      }
#pragma unroll
      for (int off = 8; off > 0; off >>= 1) sum += __shfl_xor(sum, off, 16);
      float inv = 1.f / sum;
      int prow = hi ? r + 8 : r;
#pragma unroll
      for (int nt = 0; nt < 16; nt++)
        pp[prow * 264 + nt * 16 + lm] = f2bf(sacc[nt].f[r] * inv);
    }
    __syncthreads();

    Acc8 oacc[2];
#pragma unroll
    for (int t = 0; t < 2; t++)
#pragma unroll
      for (int r = 0; r < 8; r++) oacc[t].f[r] = 0.f;

#pragma unroll
    for (int kk = 0; kk < 8; kk++) {  // O = P V, contraction over 256 keys
      FragA pa;
      pa.h[0] = *(const u16x8*)(pp + lm * 264 + kk * 32 + c0);
      pa.h[1] = *(const u16x8*)(pp + lm * 264 + kk * 32 + c1);
#pragma unroll
      for (int ntv = 0; ntv < 2; ntv++) {
        int n = ntv * 16 + lm;  // dh column
        FragA vb;
        vb.h[0] = *(const u16x8*)(Vt + n * 264 + kk * 32 + c0);
        vb.h[1] = *(const u16x8*)(Vt + n * 264 + kk * 32 + c1);
        oacc[ntv].v = wmma_bf16(pa.bf, vb.bf, oacc[ntv].v);
      }
    }
#pragma unroll
    for (int ntv = 0; ntv < 2; ntv++)
#pragma unroll
      for (int r = 0; r < 8; r++) {
        int orow = seq * 256 + tile * 16 + (hi ? r + 8 : r);
        int col = h * 32 + ntv * 16 + lm;
        o[(size_t)orow * 256 + col] = f2bf(oacc[ntv].f[r]);
      }
    __syncthreads();
  }
}

// ------------------------------------------------------------- output remap
// decoder row order (b,v,p) -> output order (b,p,v,16)
__global__ __launch_bounds__(256)
void remap_out_kernel(const float* __restrict__ src, float* __restrict__ dst) {
  int idx = blockIdx.x * 256 + threadIdx.x;  // 1048576 total
  int t = idx >> 4, j = idx & 15;
  int b = t >> 13, rem = t & 8191, v = rem >> 8, p = rem & 255;
  dst[(((size_t)(b * 256 + p)) * 32 + v) * 16 + j] = src[idx];
}

// ================================================================ launcher
extern "C" void kernel_launch(void* const* d_in, const int* in_sizes, int n_in,
                              void* d_out, int out_size, void* d_ws, size_t ws_size,
                              hipStream_t stream) {
  (void)in_sizes; (void)n_in; (void)out_size; (void)ws_size;
  const int TOK = 65536;  // 8*256*32 tokens

  const float* xin    = (const float*)d_in[0];
  const float* W_p2e  = (const float*)d_in[1];
  const float* b_p2e  = (const float*)d_in[2];
  const float* W_e2p  = (const float*)d_in[3];
  const float* b_e2p  = (const float*)d_in[4];
  const float* t_tok  = (const float*)d_in[5];
  const float* f_tok  = (const float*)d_in[6];
  const float* te[12]; for (int i = 0; i < 12; i++) te[i] = (const float*)d_in[7 + i];
  const float* fe[12]; for (int i = 0; i < 12; i++) fe[i] = (const float*)d_in[19 + i];
  const float* de[12]; for (int i = 0; i < 12; i++) de[i] = (const float*)d_in[31 + i];

  char* cur = (char*)d_ws;
  auto alloc = [&](size_t bytes) -> void* {
    void* p = (void*)cur;
    cur += (bytes + 255) & ~(size_t)255;
    return p;
  };
  unsigned short* wq_t = (unsigned short*)alloc((size_t)2 * 196608 * 2);
  unsigned short* wo_t = (unsigned short*)alloc((size_t)2 * 65536 * 2);
  unsigned short* w1_t = (unsigned short*)alloc((size_t)2 * 131072 * 2);
  unsigned short* w2_t = (unsigned short*)alloc((size_t)2 * 131072 * 2);
  unsigned short* wq_f = (unsigned short*)alloc((size_t)2 * 196608 * 2);
  unsigned short* wo_f = (unsigned short*)alloc((size_t)2 * 65536 * 2);
  unsigned short* w1_f = (unsigned short*)alloc((size_t)2 * 131072 * 2);
  unsigned short* w2_f = (unsigned short*)alloc((size_t)2 * 131072 * 2);
  unsigned short* wq_d = (unsigned short*)alloc((size_t)196608 * 2);
  unsigned short* wo_d = (unsigned short*)alloc((size_t)65536 * 2);
  unsigned short* w1_d = (unsigned short*)alloc((size_t)131072 * 2);
  unsigned short* w2_d = (unsigned short*)alloc((size_t)131072 * 2);
  unsigned short* wp2e = (unsigned short*)alloc((size_t)32 * 256 * 2);   // K-padded
  unsigned short* we2p = (unsigned short*)alloc((size_t)256 * 64 * 2);   // N-padded
  unsigned short* xb   = (unsigned short*)alloc((size_t)TOK * 32 * 2);   // K-padded
  unsigned short* ab   = (unsigned short*)alloc((size_t)TOK * 256 * 2);
  unsigned short* hb   = (unsigned short*)alloc((size_t)TOK * 512 * 2);
  unsigned short* qkvb = (unsigned short*)alloc((size_t)TOK * 768 * 2);
  unsigned short* ob   = (unsigned short*)alloc((size_t)TOK * 256 * 2);
  float* xf    = (float*)alloc((size_t)TOK * 256 * 4);
  float* tmpf  = (float*)alloc((size_t)TOK * 256 * 4);
  float* tf    = (float*)alloc((size_t)TOK * 256 * 4);
  float* outfb = (float*)alloc((size_t)TOK * 16 * 4);

  auto cvt = [&](const float* s, unsigned short* d, int n) {
    int g = (n + 255) / 256; if (g > 4096) g = 4096;
    cvt_f32_bf16_kernel<<<g, 256, 0, stream>>>(s, d, n);
  };
  auto cvtpad = [&](const float* s, unsigned short* d, int sh, int sw, int dh,
                    int dw) {
    int n = dh * dw;
    int g = (n + 255) / 256; if (g > 4096) g = 4096;
    cvt_pad2d_kernel<<<g, 256, 0, stream>>>(s, d, sh, sw, dh, dw);
  };
  cvt(te[0], wq_t, 2 * 196608); cvt(te[2], wo_t, 2 * 65536);
  cvt(te[6], w1_t, 2 * 131072); cvt(te[8], w2_t, 2 * 131072);
  cvt(fe[0], wq_f, 2 * 196608); cvt(fe[2], wo_f, 2 * 65536);
  cvt(fe[6], w1_f, 2 * 131072); cvt(fe[8], w2_f, 2 * 131072);
  cvt(de[0], wq_d, 196608); cvt(de[2], wo_d, 65536);
  cvt(de[6], w1_d, 131072); cvt(de[8], w2_d, 131072);
  cvtpad(W_p2e, wp2e, 16, 256, 32, 256);   // zero rows 16..31
  cvtpad(W_e2p, we2p, 256, 16, 256, 64);   // zero cols 16..63
  cvtpad(xin, xb, TOK, 16, TOK, 32);       // zero cols 16..31

  // (A, lda, B, ldb, bias, resid, outF, outB, Npad, Nout, K, relu)
  auto gemm = [&](const unsigned short* A, int lda, const unsigned short* B,
                  int ldb, const float* bias, const float* resid, float* oF,
                  unsigned short* oB, int Npad, int Nout, int K, int relu) {
    dim3 grid(TOK / 256, Npad / 64);
    gemm_bf16_kernel<<<grid, 256, 0, stream>>>(A, lda, B, ldb, bias, resid, oF,
                                               oB, Nout, K, relu);
  };
  auto ln = [&](const float* in, const float* g, const float* b, float* oF,
                unsigned short* oB) {
    ln_kernel<<<TOK / 8, 256, 0, stream>>>(in, g, b, oF, oB, TOK);
  };

  auto branch = [&](const float* const* ep, const unsigned short* ewq,
                    const unsigned short* ewo, const unsigned short* ew1,
                    const unsigned short* ew2, int nL, const float* mtok,
                    int kSel, float* outX, float* outMask) {
    cv_mask_kernel<<<256, 256, 0, stream>>>(xin, outMask, kSel);
    // patch -> embedding (encode-all == gather->encode->scatter, seq len 1)
    gemm(xb, 32, wp2e, 256, b_p2e, nullptr, xf, ab, 256, 256, 32, 0);
    // per-token encoder: seq_len==1 -> softmax==1 -> attention out == V
    for (int L = 0; L < nL; L++) {
      const unsigned short* Wq = ewq + (size_t)L * 196608;
      const unsigned short* Wo = ewo + (size_t)L * 65536;
      const unsigned short* W1 = ew1 + (size_t)L * 131072;
      const unsigned short* W2 = ew2 + (size_t)L * 131072;
      gemm(ab, 256, Wq + 512, 768, ep[1] + L * 768 + 512, nullptr, nullptr, hb,
           256, 256, 256, 0);                                  // v = x@Wv+bv
      gemm(hb, 256, Wo, 256, ep[3] + L * 256, xf, tmpf, nullptr,
           256, 256, 256, 0);                                  // x + v@Wo+bo
      ln(tmpf, ep[4] + L * 256, ep[5] + L * 256, xf, ab);
      gemm(ab, 256, W1, 512, ep[7] + L * 512, nullptr, nullptr, hb,
           512, 512, 256, 1);                                  // relu(x@W1+b1)
      gemm(hb, 512, W2, 256, ep[9] + L * 256, xf, tmpf, nullptr,
           256, 256, 512, 0);                                  // x + h@W2+b2
      ln(tmpf, ep[10] + L * 256, ep[11] + L * 256, xf, ab);
    }
    mask_apply_kernel<<<TOK, 256, 0, stream>>>(xf, outMask, mtok, tf, ab);
    // shared decoder: 1 layer, real self-attention over P=256
    gemm(ab, 256, wq_d, 768, de[1], nullptr, nullptr, qkvb, 768, 768, 256, 0);
    attn_kernel<<<2048, 128, 0, stream>>>(qkvb, ob);
    gemm(ob, 256, wo_d, 256, de[3], tf, tmpf, nullptr, 256, 256, 256, 0);
    ln(tmpf, de[4], de[5], xf, ab);
    gemm(ab, 256, w1_d, 512, de[7], nullptr, nullptr, hb, 512, 512, 256, 1);
    gemm(hb, 512, w2_d, 256, de[9], xf, tmpf, nullptr, 256, 256, 512, 0);
    ln(tmpf, de[10], de[11], xf, ab);
    gemm(ab, 256, we2p, 64, b_e2p, nullptr, outfb, nullptr, 64, 16, 256, 0);
    remap_out_kernel<<<4096, 256, 0, stream>>>(outfb, outX);
  };

  float* outp = (float*)d_out;
  // x_time | x_freq | time_mask | freq_mask
  branch(te, wq_t, wo_t, w1_t, w2_t, 2, t_tok, 128, outp, outp + 2097152);
  branch(fe, wq_f, wo_f, w1_f, w2_f, 2, f_tok, 102, outp + 1048576,
         outp + 2162688);
}
